// Attndecoder_75548474737071
// MI455X (gfx1250) — matrix operand order repaired and checked by reading, not compiled, not run on hardware
//
#include <hip/hip_runtime.h>

// ---------------- problem dims ----------------
#define B_   128
#define H_   512
#define EMB_ 256
#define LT_  512
#define LA_  128
#define L_   640
#define VOC_ 50000
#define EXT_ 60000
#define H2_  1024
#define H3_  1536

typedef __attribute__((ext_vector_type(16))) __bf16 v16bf;
typedef __attribute__((ext_vector_type(8)))  float  v8f;

// Native f32->bf16 conversion (lets the compiler emit packed bf16 cvt ops
// instead of manual integer rounding arithmetic).
__device__ __forceinline__ uint4 pack8(float4 a, float4 b) {
  union { __bf16 h[8]; uint4 u; } r;
  r.h[0] = (__bf16)a.x; r.h[1] = (__bf16)a.y; r.h[2] = (__bf16)a.z; r.h[3] = (__bf16)a.w;
  r.h[4] = (__bf16)b.x; r.h[5] = (__bf16)b.y; r.h[6] = (__bf16)b.z; r.h[7] = (__bf16)b.w;
  return r.u;
}
__device__ __forceinline__ float sigmoid_(float x)  { return 1.f / (1.f + __expf(-x)); }
__device__ __forceinline__ float softplus_(float x) { return (x > 20.f) ? x : log1pf(__expf(x)); }

// =====================================================================
// Generic bf16-WMMA GEMM:  C[M,N] = act( A[M,K] * W[N,K]^T + bias + addVec[r%addMod] )
// mode 0: store C.   mode 1: e[r] (+)= sum_n act(...)*redW[n]  (+ redB once)
// Block: 128 threads = 4 waves, 64x64 tile, K-stage 64 (2 k-halves -> 8 WMMAs
// per barrier pair). M must be mult of 64, K mult of 64; N guarded (N=50000).
// LDS tiles are pre-swizzled into the ISA fragment layout so each lane reads
// its whole 16-element bf16 fragment as one contiguous 32B LDS load.
//   A frag (16x32):  lane = (m&15) | ((kgrp&1)<<4), elem = (k&7) + (k>=16?8:0)
//   B frag (32x16):  lane = (n&15) | ((kgrp&1)<<4), elem = (k&7) + (k>=16?8:0)
// =====================================================================
__global__ void __launch_bounds__(128)
wmma_gemm_kernel(const float* __restrict__ A, int lda,
                 const float* __restrict__ W,
                 const float* __restrict__ bias,
                 const float* __restrict__ addVec, int addMod,
                 float* __restrict__ C, int ldc,
                 int M, int N, int K, int act, int mode,
                 const float* __restrict__ redW,
                 const float* __restrict__ redB,
                 float* __restrict__ redOut)
{
  __shared__ __attribute__((aligned(16))) unsigned short lA[2][4][32][16];
  __shared__ __attribute__((aligned(16))) unsigned short lB[2][4][32][16];
  __shared__ float eAcc[64];

  const int tid    = threadIdx.x;
  const int lane   = tid & 31;
  const int wid    = tid >> 5;
  const int mBlock = blockIdx.y * 64;
  const int nBlock = blockIdx.x * 64;
  const int mFrag  = (wid >> 1) << 1;   // 0 or 2
  const int nFrag  = (wid & 1) << 1;    // 0 or 2

  if (mode == 1 && tid < 64) eAcc[tid] = 0.f;

  v8f acc[2][2] = {};

  for (int k0 = 0; k0 < K; k0 += 64) {
#pragma unroll
    for (int u = 0; u < 4; ++u) {
      int unit = tid + u * 128;           // 512 (row, kgroup) units
      int row  = unit >> 3;               // 0..63
      int g    = unit & 7;                // k-group of 8 within the 64-wide stage
      int kh   = g >> 2;                  // which k-half (two 16x32 / 32x16 frags)
      int gg   = g & 3;
      { // ---- A tile: rows always in range ----
        const float* src = A + (size_t)(mBlock + row) * lda + (k0 + g * 8);
        float4 f0 = *(const float4*)(src);
        float4 f1 = *(const float4*)(src + 4);
        int la = (row & 15) | ((gg & 1) << 4);
        int eb = (gg >> 1) << 3;
        *(uint4*)&lA[kh][row >> 4][la][eb] = pack8(f0, f1);
      }
      { // ---- B tile from W[N,K] (guard N) ----
        int n = nBlock + row;
        float4 f0 = {0.f, 0.f, 0.f, 0.f}, f1 = {0.f, 0.f, 0.f, 0.f};
        if (n < N) {
          const float* src = W + (size_t)n * K + (k0 + g * 8);
          f0 = *(const float4*)(src);
          f1 = *(const float4*)(src + 4);
        }
        int la = (row & 15) | ((gg & 1) << 4);
        int eb = (gg >> 1) << 3;
        *(uint4*)&lB[kh][row >> 4][la][eb] = pack8(f0, f1);
      }
    }
    if (k0 + 64 < K)  // speculative prefetch of next A K-stage (global_prefetch_b8)
      __builtin_prefetch(A + (size_t)(mBlock + (tid >> 1)) * lda + (k0 + 64), 0, 1);
    __syncthreads();

#pragma unroll
    for (int kh = 0; kh < 2; ++kh) {
      v16bf af[2], bw[2];
      af[0] = *(const v16bf*)&lA[kh][mFrag + 0][lane][0];
      af[1] = *(const v16bf*)&lA[kh][mFrag + 1][lane][0];
      bw[0] = *(const v16bf*)&lB[kh][nFrag + 0][lane][0];
      bw[1] = *(const v16bf*)&lB[kh][nFrag + 1][lane][0];
#pragma unroll
      for (int i = 0; i < 2; ++i)
#pragma unroll
        for (int j = 0; j < 2; ++j)
          acc[i][j] = __builtin_amdgcn_wmma_f32_16x16x32_bf16(
              false, af[i], false, bw[j], (short)0, acc[i][j], false, false);
    }
    __syncthreads();
  }

  // ---- epilogue ----
  const int rowOff = (lane >> 4) << 3;  // C layout: vgpr v -> m = v + (lane>=16)*8
  const int nCol   = lane & 15;
#pragma unroll
  for (int i = 0; i < 2; ++i) {
#pragma unroll
    for (int j = 0; j < 2; ++j) {
      int  nG  = nBlock + ((nFrag + j) << 4) + nCol;
      bool nOK = nG < N;
      float bv = (bias && nOK) ? bias[nG] : 0.f;
      float wv = (mode == 1 && nOK) ? redW[nG] : 0.f;
#pragma unroll
      for (int v = 0; v < 8; ++v) {
        int mG = mBlock + ((mFrag + i) << 4) + rowOff + v;
        float val = acc[i][j][v] + bv;
        if (addVec && nOK) val += addVec[(size_t)(mG % addMod) * N + nG];
        if (act == 1)      val = softplus_(val);
        else if (act == 2) val = tanhf(val);
        if (mode == 0) {
          if (nOK) C[(size_t)mG * ldc + nG] = val;
        } else {
          atomicAdd(&eAcc[mG - mBlock], val * wv);   // ds_add_f32
        }
      }
    }
  }
  if (mode == 1) {
    __syncthreads();
    if (tid < 64) {
      float v = eAcc[tid];
      if (blockIdx.x == 0 && redB) v += redB[0];
      atomicAdd(&redOut[mBlock + tid], v);
    }
  }
}

// ---------------- GRU elementwise: h' = (1-z)*n + z*h ----------------
__global__ void gru_elem_kernel(const float* __restrict__ gi, const float* __restrict__ gh,
                                const float* __restrict__ hprev, int hStride,
                                float* __restrict__ hout, int oStride)
{
  int b = blockIdx.x;
  int j = blockIdx.y * 256 + threadIdx.x;   // 0..511
  const float* gib = gi + (size_t)b * H3_;
  const float* ghb = gh + (size_t)b * H3_;
  float r = sigmoid_(gib[j] + ghb[j]);
  float z = sigmoid_(gib[H_ + j] + ghb[H_ + j]);
  float n = tanhf(gib[2 * H_ + j] + r * ghb[2 * H_ + j]);
  float h = hprev[(size_t)b * hStride + j];
  hout[(size_t)b * oStride + j] = (1.f - z) * n + z * h;
}

// ---------------- softmax over l (e stored [l*B + b]) -> a[b, l] ----------------
__global__ void col_softmax_kernel(const float* __restrict__ e, int Lr, float* __restrict__ a)
{
  __shared__ float sm[256];
  int b = blockIdx.x, tid = threadIdx.x;
  float m = -3.0e38f;
  for (int l = tid; l < Lr; l += 256) m = fmaxf(m, e[(size_t)l * B_ + b]);
  sm[tid] = m; __syncthreads();
  for (int s = 128; s > 0; s >>= 1) { if (tid < s) sm[tid] = fmaxf(sm[tid], sm[tid + s]); __syncthreads(); }
  m = sm[0]; __syncthreads();
  float sum = 0.f;
  for (int l = tid; l < Lr; l += 256) sum += __expf(e[(size_t)l * B_ + b] - m);
  sm[tid] = sum; __syncthreads();
  for (int s = 128; s > 0; s >>= 1) { if (tid < s) sm[tid] += sm[tid + s]; __syncthreads(); }
  float inv = 1.f / sm[0];
  for (int l = tid; l < Lr; l += 256)
    a[(size_t)b * Lr + l] = __expf(e[(size_t)l * B_ + b] - m) * inv;
}

// ---------------- ctx[b,d] = sum_l a[b,l] * outs[l,b,d] ----------------
__global__ void ctx_kernel(const float* __restrict__ a, const float* __restrict__ outs,
                           int Lr, float* __restrict__ ctx)
{
  int b = blockIdx.x;
  int d = blockIdx.y * 256 + threadIdx.x;   // 0..1023
  float s = 0.f;
  for (int l = 0; l < Lr; ++l)
    s += a[(size_t)b * Lr + l] * outs[((size_t)l * B_ + b) * H2_ + d];
  ctx[(size_t)b * H2_ + d] = s;
}

// ---------------- 2-way softmax weights ----------------
__global__ void weights2_kernel(const float* __restrict__ e2, float* __restrict__ wts)
{
  int b = threadIdx.x;
  float e0 = e2[b], e1 = e2[B_ + b];
  float m = fmaxf(e0, e1);
  float x0 = __expf(e0 - m), x1 = __expf(e1 - m);
  float inv = 1.f / (x0 + x1);
  wts[b] = x0 * inv; wts[B_ + b] = x1 * inv;
}

// ---------------- context[b,d] = w0*rep0 + w1*rep1 ----------------
__global__ void context_kernel(const float* __restrict__ wts, const float* __restrict__ rep,
                               float* __restrict__ ctxv)
{
  int b = blockIdx.x;
  int d = blockIdx.y * 256 + threadIdx.x;
  ctxv[(size_t)b * H2_ + d] = wts[b] * rep[(size_t)b * H2_ + d]
                            + wts[B_ + b] * rep[(size_t)B_ * H2_ + (size_t)b * H2_ + d];
}

// ---------------- attn_dist[b,l] ----------------
__global__ void attn_dist_kernel(const float* __restrict__ wts, const float* __restrict__ a_t,
                                 const float* __restrict__ a_a, float* __restrict__ out)
{
  int b = blockIdx.x;
  float w0 = wts[b], w1 = wts[B_ + b];
  for (int l = threadIdx.x; l < L_; l += 256)
    out[(size_t)b * L_ + l] = (l < LT_) ? w0 * a_t[(size_t)b * LT_ + l]
                                        : w1 * a_a[(size_t)b * LA_ + (l - LT_)];
}

// ---------------- p_gens ----------------
__global__ void pgen_kernel(const float* __restrict__ ctxv, const float* __restrict__ S,
                            const float* __restrict__ x,
                            const float* __restrict__ wh_w, const float* __restrict__ wh_b,
                            const float* __restrict__ ws_w, const float* __restrict__ ws_b,
                            const float* __restrict__ wx_w, const float* __restrict__ wx_b,
                            float* __restrict__ pg)
{
  int b = threadIdx.x;
  float s = wh_b[0] + ws_b[0] + wx_b[0];
  for (int d = 0; d < H2_;  ++d) s += ctxv[(size_t)b * H2_ + d] * wh_w[d];
  for (int d = 0; d < H_;   ++d) s += S[(size_t)b * H_ + d]    * ws_w[d];
  for (int d = 0; d < EMB_; ++d) s += x[(size_t)b * EMB_ + d]  * wx_w[d];
  pg[b] = sigmoid_(s);
}

// ---------------- cat = [S | context] ----------------
__global__ void cat_kernel(const float* __restrict__ S, const float* __restrict__ ctxv,
                           float* __restrict__ cat)
{
  int b = blockIdx.x;
  int j = blockIdx.y * 256 + threadIdx.x;   // 0..1535
  cat[(size_t)b * H3_ + j] = (j < H_) ? S[(size_t)b * H_ + j]
                                      : ctxv[(size_t)b * H2_ + (j - H_)];
}

// ---------------- vocab softmax * p_gen -> vocab_ext (padded) ----------------
__global__ void vocab_softmax_kernel(const float* __restrict__ Z2, const float* __restrict__ pg,
                                     float* __restrict__ vext)
{
  __shared__ float sm[256];
  int b = blockIdx.x, tid = threadIdx.x;
  const float* row = Z2 + (size_t)b * VOC_;
  float m = -3.0e38f;
  for (int j = tid; j < VOC_; j += 256) m = fmaxf(m, row[j]);
  sm[tid] = m; __syncthreads();
  for (int s = 128; s > 0; s >>= 1) { if (tid < s) sm[tid] = fmaxf(sm[tid], sm[tid + s]); __syncthreads(); }
  m = sm[0]; __syncthreads();
  float sum = 0.f;
  for (int j = tid; j < VOC_; j += 256) sum += __expf(row[j] - m);
  sm[tid] = sum; __syncthreads();
  for (int s = 128; s > 0; s >>= 1) { if (tid < s) sm[tid] += sm[tid + s]; __syncthreads(); }
  float scale = pg[b] / sm[0];
  for (int j = tid; j < EXT_; j += 256)
    vext[(size_t)b * EXT_ + j] = (j < VOC_) ? __expf(row[j] - m) * scale : 0.f;
}

// ---------------- masked renormalize of attn_dist ----------------
__global__ void renorm_kernel(const float* __restrict__ attn, const float* __restrict__ mask,
                              float* __restrict__ ren)
{
  __shared__ float sm[256];
  int b = blockIdx.x, tid = threadIdx.x;
  float sum = 0.f;
  for (int l = tid; l < L_; l += 256) {
    float v = attn[(size_t)b * L_ + l] * mask[(size_t)b * L_ + l];
    ren[(size_t)b * L_ + l] = v;
    sum += v;
  }
  sm[tid] = sum; __syncthreads();
  for (int s = 128; s > 0; s >>= 1) { if (tid < s) sm[tid] += sm[tid + s]; __syncthreads(); }
  float inv = 1.f / sm[0];
  for (int l = tid; l < L_; l += 256) ren[(size_t)b * L_ + l] *= inv;
}

// ---------------- scatter-add into extended vocab ----------------
__global__ void scatter_kernel(const int* __restrict__ data, const float* __restrict__ ren,
                               float* __restrict__ proj)
{
  int b = blockIdx.x;
  for (int i = threadIdx.x; i < L_; i += 256) {
    int idx = data[(size_t)b * L_ + i];
    atomicAdd(&proj[(size_t)b * EXT_ + idx], ren[(size_t)b * L_ + i]);
  }
}

// ---------------- final = softmax(vocab_ext + (1-pg)*proj) ----------------
__global__ void final_softmax_kernel(const float* __restrict__ vext, const float* __restrict__ proj,
                                     const float* __restrict__ pg, float* __restrict__ fin)
{
  __shared__ float sm[256];
  int b = blockIdx.x, tid = threadIdx.x;
  float sc = 1.f - pg[b];
  const float* ve = vext + (size_t)b * EXT_;
  const float* pr = proj + (size_t)b * EXT_;
  float m = -3.0e38f;
  for (int j = tid; j < EXT_; j += 256) m = fmaxf(m, ve[j] + sc * pr[j]);
  sm[tid] = m; __syncthreads();
  for (int s = 128; s > 0; s >>= 1) { if (tid < s) sm[tid] = fmaxf(sm[tid], sm[tid + s]); __syncthreads(); }
  m = sm[0]; __syncthreads();
  float sum = 0.f;
  for (int j = tid; j < EXT_; j += 256) sum += __expf(ve[j] + sc * pr[j] - m);
  sm[tid] = sum; __syncthreads();
  for (int s = 128; s > 0; s >>= 1) { if (tid < s) sm[tid] += sm[tid + s]; __syncthreads(); }
  float inv = 1.f / sm[0];
  for (int j = tid; j < EXT_; j += 256)
    fin[(size_t)b * EXT_ + j] = __expf(ve[j] + sc * pr[j] - m) * inv;
}

__global__ void zero_kernel(float* __restrict__ p, int n)
{
  int i = blockIdx.x * 256 + threadIdx.x;
  if (i < n) p[i] = 0.f;
}

// =====================================================================
// host-side launcher
// =====================================================================
static void gemm(hipStream_t s, const float* A, int lda, const float* W,
                 const float* bias, const float* addVec, int addMod,
                 float* C, int ldc, int M, int N, int K, int act,
                 int mode = 0, const float* redW = nullptr,
                 const float* redB = nullptr, float* redOut = nullptr)
{
  dim3 grid((N + 63) / 64, M / 64);
  wmma_gemm_kernel<<<grid, 128, 0, s>>>(A, lda, W, bias, addVec, addMod,
                                        C, ldc, M, N, K, act, mode, redW, redB, redOut);
}
static void zero(hipStream_t s, float* p, int n)
{
  zero_kernel<<<(n + 255) / 256, 256, 0, s>>>(p, n);
}

extern "C" void kernel_launch(void* const* d_in, const int* in_sizes, int n_in,
                              void* d_out, int out_size, void* d_ws, size_t ws_size,
                              hipStream_t stream)
{
  (void)in_sizes; (void)n_in; (void)out_size; (void)ws_size;
  // ---- inputs (setup_inputs order; params dict flattened in insertion order) ----
  const int*   data   = (const int*)d_in[0];
  const float* x      = (const float*)d_in[1];   // target_input [1,B,EMB]
  const float* mask   = (const float*)d_in[2];
  const float* hidden = (const float*)d_in[3];   // [1,B,H]
  const float* outs_t = (const float*)d_in[4];   // [LT,B,2H]
  const float* outs_a = (const float*)d_in[5];   // [LA,B,2H]
  const float* hc     = (const float*)d_in[6];   // [2,B,H]
  int pi = 7;
  auto F = [&](void) { return (const float*)d_in[pi++]; };
  const float *Wh_t_w=F(), *Wh_t_b=F(), *Ws_t_w=F(), *Ws_t_b=F(), *v_t_w=F(), *v_t_b=F();
  const float *Wh_a_w=F(), *Wh_a_b=F(), *Ws_a_w=F(), *Ws_a_b=F(), *v_a_w=F(), *v_a_b=F();
  const float *wt_w=F(),  *wt_b=F(),  *ws2_w=F(),  *ws2_b=F(),  *v_w=F(),   *v_b=F();
  const float *V1_w=F(),  *V1_b=F(),  *V2_w=F(),   *V2_b=F();
  const float *wh_w=F(),  *wh_b=F(),  *wsl_w=F(),  *wsl_b=F(),  *wx_w=F(),  *wx_b=F();
  const float *gruWih=F(), *gruWhh=F(), *gruBih=F(), *gruBhh=F();
  const float *gvfWih=F(), *gvfWhh=F(), *gvfBih=F(), *gvfBhh=F();
  const float *gvbWih=F(), *gvbWhh=F(), *gvbBih=F(), *gvbBhh=F();

  // ---- output regions (tuple order: S, attn_dist, p_gens, final, vocab_ext) ----
  float* out      = (float*)d_out;
  float* S_out    = out;
  float* attn_out = out + (size_t)B_ * H_;
  float* pg_out   = attn_out + (size_t)B_ * L_;
  float* fin_out  = pg_out + B_;
  float* vext_out = fin_out + (size_t)B_ * EXT_;

  // ---- workspace ----
  float* wsp = (float*)d_ws;
  size_t o = 0;
  float* gi   = wsp + o; o += (size_t)B_ * H3_;
  float* gh   = wsp + o; o += (size_t)B_ * H3_;
  float* T1t  = wsp + o; o += (size_t)B_ * H_;
  float* T1a  = wsp + o; o += (size_t)B_ * H_;
  float* e_t  = wsp + o; o += (size_t)LT_ * B_;
  float* e_a  = wsp + o; o += (size_t)LA_ * B_;
  float* a_t  = wsp + o; o += (size_t)B_ * LT_;
  float* a_a  = wsp + o; o += (size_t)B_ * LA_;
  float* ctx  = wsp + o; o += (size_t)2 * B_ * H2_;   // ctx_t | ctx_a
  float* rep  = wsp + o; o += (size_t)2 * B_ * H2_;   // [2,B,2H]
  float* T2   = wsp + o; o += (size_t)B_ * H_;
  float* e2   = wsp + o; o += (size_t)2 * B_;
  float* wts  = wsp + o; o += (size_t)2 * B_;
  float* ctxv = wsp + o; o += (size_t)B_ * H2_;
  float* cat  = wsp + o; o += (size_t)B_ * H3_;
  float* Z1   = wsp + o; o += (size_t)B_ * H2_;
  float* Z2   = wsp + o; o += (size_t)B_ * VOC_;
  float* ren  = wsp + o; o += (size_t)B_ * L_;
  float* proj = wsp + o; o += (size_t)B_ * EXT_;

  // ---- 1. decoder GRU -> S ----
  gemm(stream, x,      EMB_, gruWih, gruBih, nullptr, 0, gi, H3_, B_, H3_, EMB_, 0);
  gemm(stream, hidden, H_,   gruWhh, gruBhh, nullptr, 0, gh, H3_, B_, H3_, H_,   0);
  gru_elem_kernel<<<dim3(B_, 2), 256, 0, stream>>>(gi, gh, hidden, H_, S_out, H_);

  // ---- 2. dual attention (big GEMMs with fused softplus + dot-with-v epilogue) ----
  gemm(stream, S_out, H_, Ws_t_w, Ws_t_b, nullptr, 0, T1t, H_, B_, H_, H_, 0);
  gemm(stream, S_out, H_, Ws_a_w, Ws_a_b, nullptr, 0, T1a, H_, B_, H_, H_, 0);
  zero(stream, e_t, LT_ * B_);
  zero(stream, e_a, LA_ * B_);
  gemm(stream, outs_t, H2_, Wh_t_w, Wh_t_b, T1t, B_, nullptr, 0,
       LT_ * B_, H_, H2_, /*softplus*/1, /*reduce*/1, v_t_w, v_t_b, e_t);
  gemm(stream, outs_a, H2_, Wh_a_w, Wh_a_b, T1a, B_, nullptr, 0,
       LA_ * B_, H_, H2_, 1, 1, v_a_w, v_a_b, e_a);
  col_softmax_kernel<<<B_, 256, 0, stream>>>(e_t, LT_, a_t);
  col_softmax_kernel<<<B_, 256, 0, stream>>>(e_a, LA_, a_a);
  ctx_kernel<<<dim3(B_, 4), 256, 0, stream>>>(a_t, outs_t, LT_, ctx);
  ctx_kernel<<<dim3(B_, 4), 256, 0, stream>>>(a_a, outs_a, LA_, ctx + (size_t)B_ * H2_);

  // ---- 3. bidirectional gru_v over [ctx_t, ctx_a] -> rep ----
  const float* hc0 = hc;
  const float* hc1 = hc + (size_t)B_ * H_;
  float* h1 = rep;                               // rep[0][:, :H]
  float* h2 = rep + (size_t)B_ * H2_;            // rep[1][:, :H]
  float* g1 = rep + (size_t)B_ * H2_ + H_;       // rep[1][:, H:]
  float* g2 = rep + H_;                          // rep[0][:, H:]
  const float* ctx_t = ctx;
  const float* ctx_a = ctx + (size_t)B_ * H2_;
  // h1 = cell_f(ctx_t, hc0)
  gemm(stream, ctx_t, H2_, gvfWih, gvfBih, nullptr, 0, gi, H3_, B_, H3_, H2_, 0);
  gemm(stream, hc0,   H_,  gvfWhh, gvfBhh, nullptr, 0, gh, H3_, B_, H3_, H_,  0);
  gru_elem_kernel<<<dim3(B_, 2), 256, 0, stream>>>(gi, gh, hc0, H_, h1, H2_);
  // h2 = cell_f(ctx_a, h1)
  gemm(stream, ctx_a, H2_, gvfWih, gvfBih, nullptr, 0, gi, H3_, B_, H3_, H2_, 0);
  gemm(stream, h1,    H2_, gvfWhh, gvfBhh, nullptr, 0, gh, H3_, B_, H3_, H_,  0);
  gru_elem_kernel<<<dim3(B_, 2), 256, 0, stream>>>(gi, gh, h1, H2_, h2, H2_);
  // g1 = cell_b(ctx_a, hc1)
  gemm(stream, ctx_a, H2_, gvbWih, gvbBih, nullptr, 0, gi, H3_, B_, H3_, H2_, 0);
  gemm(stream, hc1,   H_,  gvbWhh, gvbBhh, nullptr, 0, gh, H3_, B_, H3_, H_,  0);
  gru_elem_kernel<<<dim3(B_, 2), 256, 0, stream>>>(gi, gh, hc1, H_, g1, H2_);
  // g2 = cell_b(ctx_t, g1)
  gemm(stream, ctx_t, H2_, gvbWih, gvbBih, nullptr, 0, gi, H3_, B_, H3_, H2_, 0);
  gemm(stream, g1,    H2_, gvbWhh, gvbBhh, nullptr, 0, gh, H3_, B_, H3_, H_,  0);
  gru_elem_kernel<<<dim3(B_, 2), 256, 0, stream>>>(gi, gh, g1, H2_, g2, H2_);

  // ---- 4. meta-attention over rep ----
  gemm(stream, S_out, H_, ws2_w, ws2_b, nullptr, 0, T2, H_, B_, H_, H_, 0);
  zero(stream, e2, 2 * B_);
  gemm(stream, rep, H2_, wt_w, wt_b, T2, B_, nullptr, 0,
       2 * B_, H_, H2_, /*tanh*/2, /*reduce*/1, v_w, v_b, e2);
  weights2_kernel<<<1, 128, 0, stream>>>(e2, wts);
  context_kernel<<<dim3(B_, 4), 256, 0, stream>>>(wts, rep, ctxv);
  attn_dist_kernel<<<B_, 256, 0, stream>>>(wts, a_t, a_a, attn_out);
  pgen_kernel<<<1, 128, 0, stream>>>(ctxv, S_out, x, wh_w, wh_b, wsl_w, wsl_b, wx_w, wx_b, pg_out);

  // ---- 5. vocab distribution ----
  cat_kernel<<<dim3(B_, 6), 256, 0, stream>>>(S_out, ctxv, cat);
  gemm(stream, cat, H3_, V1_w, V1_b, nullptr, 0, Z1, H2_,  B_, H2_,  H3_, 0);
  gemm(stream, Z1,  H2_, V2_w, V2_b, nullptr, 0, Z2, VOC_, B_, VOC_, H2_, 0);
  vocab_softmax_kernel<<<B_, 256, 0, stream>>>(Z2, pg_out, vext_out);

  // ---- 6. pointer path + final ----
  renorm_kernel<<<B_, 256, 0, stream>>>(attn_out, mask, ren);
  zero(stream, proj, B_ * EXT_);
  scatter_kernel<<<B_, 256, 0, stream>>>(data, ren, proj);
  final_softmax_kernel<<<B_, 256, 0, stream>>>(vext_out, proj, pg_out, fin_out);
}
